// IGNNConv_4664334484030
// MI455X (gfx1250) — compile-verified, matrix operand order checked by reading, not run on hardware
//
#include <hip/hip_runtime.h>
#include <hip/hip_bf16.h>
#include <math.h>

#define NNODES 100000
#define NEDGES 3200000
#define KIN    256
#define HF     128
#define NHOPS  4

typedef __attribute__((ext_vector_type(16))) __bf16 v16bf;
typedef __attribute__((ext_vector_type(8)))  float  v8f;

__device__ __forceinline__ unsigned short f2bf(float x) {
    union { float f; unsigned int u; } c; c.f = x;
    unsigned int r = c.u + 0x7FFFu + ((c.u >> 16) & 1u);   // round-to-nearest-even
    return (unsigned short)(r >> 16);
}
__device__ __forceinline__ unsigned int pack2bf(float a, float b) {
    return (unsigned int)f2bf(a) | ((unsigned int)f2bf(b) << 16);
}

// ---------------------------------------------------------------------------
// Degree / symmetric-norm precompute: dinv[n] = rsqrt(1 + indegree(n))
// ---------------------------------------------------------------------------
__global__ void k_deg_init(float* __restrict__ dinv) {
    int n = blockIdx.x * 256 + threadIdx.x;
    if (n < NNODES) dinv[n] = 1.0f;                        // self-loop
}

__global__ void k_deg_count(const int* __restrict__ dst, float* __restrict__ dinv) {
    int e = blockIdx.x * 256 + threadIdx.x;
    if (e < NEDGES) atomicAdd(&dinv[dst[e]], 1.0f);
}

__global__ void k_deg_rsqrt(float* __restrict__ dinv) {
    int n = blockIdx.x * 256 + threadIdx.x;
    if (n < NNODES) dinv[n] = rsqrtf(dinv[n]);
}

// ---------------------------------------------------------------------------
// h = relu(features @ W0 + b0)   [100000 x 256] x [256 x 128]
// Full W0 (64KB bf16, [n][k]) + A tile (8KB bf16) staged once in dynamic LDS,
// then a fully unrolled K-loop: 8x v_wmma_f32_16x16x32_bf16 per wave.
// ---------------------------------------------------------------------------
__global__ __launch_bounds__(256) void k_gemm_relu(
    const float* __restrict__ feat, const float* __restrict__ W0,
    const float* __restrict__ b0, float* __restrict__ h)
{
    extern __shared__ unsigned short smem[];
    unsigned short* lds_a = smem;              // [16][256]
    unsigned short* lds_w = smem + 16 * KIN;   // [128][256]  (transposed: [n][k])

    const int tid  = threadIdx.x;
    const int lane = tid & 31;
    const int wave = tid >> 5;
    const int row0 = blockIdx.x * 16;
    const int col0 = wave * 16;

    // --- Stage A tile: 2048 float2 pairs -> packed bf16x2, ds_store_b32 ---
    unsigned int* lds_a32 = (unsigned int*)lds_a;
    const float2* f2 = (const float2*)(feat + (size_t)row0 * KIN);
    #pragma unroll
    for (int it = 0; it < 8; ++it) {
        int p = it * 256 + tid;               // pair idx: row = p>>7, kpair = p&127
        float2 v = f2[p];                     // coalesced 8B
        lds_a32[p] = pack2bf(v.x, v.y);
    }

    // --- Stage full W0: [k][n] global -> [n][k] LDS; coalesced b32 reads ---
    unsigned int* lds_w32 = (unsigned int*)lds_w;
    {
        const int n = tid & 127;
        const int kphase = tid >> 7;          // 0 or 1
        #pragma unroll
        for (int it = 0; it < 64; ++it) {
            int k = (it * 2 + kphase) * 2;    // even k
            float a0 = W0[(size_t)k * HF + n];        // lanes sweep n: coalesced
            float a1 = W0[(size_t)(k + 1) * HF + n];
            lds_w32[(n * KIN + k) >> 1] = pack2bf(a0, a1);
        }
    }
    if (blockIdx.x + 1 < gridDim.x)           // warm L2 for next block's A tile
        __builtin_prefetch(feat + (size_t)(row0 + 16) * KIN, 0, 1);
    __syncthreads();

    // --- WMMA main loop, fully unrolled: 4x ds_load_b128 + wmma per step ---
    const int half = lane >> 4;               // 0: lanes 0-15, 1: lanes 16-31
    const int m    = lane & 15;
    const unsigned short* ar = &lds_a[m * KIN];
    const unsigned short* bc = &lds_w[(col0 + m) * KIN];

    v8f acc = {};
    #pragma unroll
    for (int kt = 0; kt < 8; ++kt) {
        // 16-bit A layout: lanes 0-15 -> K {0..7,16..23}; lanes 16-31 -> {8..15,24..31}
        union { v16bf v; unsigned short u[16]; } af, bf_;
        #pragma unroll
        for (int e = 0; e < 8; ++e) {
            af.u[e]      = ar[kt * 32 + 8 * half + e];
            af.u[8 + e]  = ar[kt * 32 + 16 + 8 * half + e];
            bf_.u[e]     = bc[kt * 32 + 8 * half + e];
            bf_.u[8 + e] = bc[kt * 32 + 16 + 8 * half + e];
        }
        acc = __builtin_amdgcn_wmma_f32_16x16x32_bf16(
                  false, af.v, false, bf_.v, (short)0, acc, false, false);
    }

    // C/D layout: VGPR j -> M = j + 8*half, N = lane&15
    const int   colg = col0 + m;
    const float bv   = b0[colg];
    #pragma unroll
    for (int j = 0; j < 8; ++j) {
        int grow = row0 + j + 8 * half;
        h[(size_t)grow * HF + colg] = fmaxf(acc[j] + bv, 0.0f);
    }
}

// ---------------------------------------------------------------------------
// agg[n,:] = h[n,:] * dinv[n]^2   (self-loop term; also initializes agg)
// ---------------------------------------------------------------------------
__global__ __launch_bounds__(256) void k_agg_init(
    const float* __restrict__ h, const float* __restrict__ dinv,
    float* __restrict__ agg)
{
    size_t idx  = (size_t)blockIdx.x * 256 + threadIdx.x;   // over N*H/4 float4s
    int    node = (int)(idx >> 5);                          // 32 float4 per node
    float  d    = dinv[node];
    float  s    = d * d;
    float4 v    = ((const float4*)h)[idx];
    v.x *= s; v.y *= s; v.z *= s; v.w *= s;
    ((float4*)agg)[idx] = v;
}

// ---------------------------------------------------------------------------
// Edge aggregation: one wave32 per edge; lane gathers one float4 (whole row =
// 32 lanes x 16B), then 4 fp32 atomics. h (51MB) is L2-resident (192MB L2).
// ---------------------------------------------------------------------------
__global__ __launch_bounds__(256) void k_edge_agg(
    const int* __restrict__ src, const int* __restrict__ dst,
    const float* __restrict__ dinv, const float* __restrict__ h,
    float* __restrict__ agg)
{
    int e    = (blockIdx.x * 256 + threadIdx.x) >> 5;       // wave32 per edge
    int lane = threadIdx.x & 31;
    if (e >= NEDGES) return;
    int   s = src[e], d = dst[e];
    float w = dinv[s] * dinv[d];
    float4 v = ((const float4*)(h + (size_t)s * HF))[lane]; // global_load_b128
    float* ad = agg + (size_t)d * HF + lane * 4;
    atomicAdd(&ad[0], v.x * w);
    atomicAdd(&ad[1], v.y * w);
    atomicAdd(&ad[2], v.z * w);
    atomicAdd(&ad[3], v.w * w);
}

// ---------------------------------------------------------------------------
// h_k = relu(agg + bias); a = tanh([h_k, h] @ Watt + batt); h = a*h_k + (1-a)*h
// One wave per node; float4 lanes + wave32 shuffle reduction for the dot.
// ---------------------------------------------------------------------------
__global__ __launch_bounds__(256) void k_combine(
    const float* __restrict__ agg, const float* __restrict__ bias,
    const float* __restrict__ watt, const float* __restrict__ batt,
    float* __restrict__ h)
{
    int node = (blockIdx.x * 256 + threadIdx.x) >> 5;
    int lane = threadIdx.x & 31;
    if (node >= NNODES) return;

    float4 av = ((const float4*)(agg + (size_t)node * HF))[lane];
    float4 hv = ((const float4*)(h   + (size_t)node * HF))[lane];
    float4 bv = ((const float4*)bias)[lane];
    float4 w1 = ((const float4*)watt)[lane];
    float4 w2 = ((const float4*)(watt + HF))[lane];

    float4 hk;
    hk.x = fmaxf(av.x + bv.x, 0.f);
    hk.y = fmaxf(av.y + bv.y, 0.f);
    hk.z = fmaxf(av.z + bv.z, 0.f);
    hk.w = fmaxf(av.w + bv.w, 0.f);

    float dot = hk.x * w1.x + hk.y * w1.y + hk.z * w1.z + hk.w * w1.w
              + hv.x * w2.x + hv.y * w2.y + hv.z * w2.z + hv.w * w2.w;
    #pragma unroll
    for (int off = 16; off > 0; off >>= 1)
        dot += __shfl_xor(dot, off, 32);

    float a = tanhf(dot + batt[0]);
    float b = 1.f - a;
    float4 o;
    o.x = a * hk.x + b * hv.x;
    o.y = a * hk.y + b * hv.y;
    o.z = a * hk.z + b * hv.z;
    o.w = a * hk.w + b * hv.w;
    ((float4*)(h + (size_t)node * HF))[lane] = o;
}

// ---------------------------------------------------------------------------
extern "C" void kernel_launch(void* const* d_in, const int* in_sizes, int n_in,
                              void* d_out, int out_size, void* d_ws, size_t ws_size,
                              hipStream_t stream) {
    const int*   edges  = (const int*)d_in[0];          // [2, E]
    const int*   src    = edges;
    const int*   dst    = edges + NEDGES;
    const float* feat   = (const float*)d_in[1];        // [N, 256]
    const float* W0     = (const float*)d_in[2];        // [256, 128]
    const float* b0     = (const float*)d_in[3];        // [128]
    const float* biases = (const float*)d_in[4];        // [4, 128]
    const float* Watt   = (const float*)d_in[5];        // [4, 256]
    const float* batt   = (const float*)d_in[6];        // [4]

    float* h    = (float*)d_out;                        // [N, 128]
    float* agg  = (float*)d_ws;                         // N*H floats
    float* dinv = agg + (size_t)NNODES * HF;            // N floats

    k_deg_init <<<(NNODES + 255) / 256, 256, 0, stream>>>(dinv);
    k_deg_count<<<(NEDGES + 255) / 256, 256, 0, stream>>>(dst, dinv);
    k_deg_rsqrt<<<(NNODES + 255) / 256, 256, 0, stream>>>(dinv);

    const size_t smem_bytes = (size_t)(16 * KIN + HF * KIN) * sizeof(unsigned short); // 72KB
    k_gemm_relu<<<NNODES / 16, 256, smem_bytes, stream>>>(feat, W0, b0, h);

    for (int i = 0; i < NHOPS; ++i) {
        k_agg_init<<<(NNODES * HF / 4) / 256, 256, 0, stream>>>(h, dinv, agg);
        k_edge_agg<<<NEDGES / 8, 256, 0, stream>>>(src, dst, dinv, h, agg);
        k_combine <<<(NNODES + 7) / 8, 256, 0, stream>>>(
            agg, biases + (size_t)i * HF, Watt + (size_t)i * 2 * HF, batt + i, h);
    }
}
